// DynamiceResidualBlock_13537736917342
// MI455X (gfx1250) — matrix operand order repaired and checked by reading, not compiled, not run on hardware
//
#include <hip/hip_runtime.h>

// Fused SNN residual block for MI455X (gfx1250, wave32):
//   GEMM1 (bf16 WMMA, K=512) -> BN1 -> LIF over T (state in acc registers)
//   -> GEMM2 (bf16 WMMA, K=64) -> BN2 -> rearranged scatter store.
// All LDS operands stored in WMMA-fragment-native order: every fragment is
// 2 aligned ds_load_b128 per lane (no scalar u16 gathers, no b16 packing).

typedef __bf16 bf16;
typedef __attribute__((ext_vector_type(16))) __bf16 v16bf;
typedef __attribute__((ext_vector_type(8)))  __bf16 v8bf;
typedef __attribute__((ext_vector_type(4)))  __bf16 v4bf;
typedef __attribute__((ext_vector_type(8)))  float  v8f;

#define T_   4
#define B_   32
#define C_   512
#define HID_ 64
#define OUT_ 32
#define N_   1024
#define NT   64           // N-tile per block
#define W1S  520          // padded w1 row stride in bf16 (bank spread)
#define XSTR 24           // bf16 per lane-slot in x fragment chunks (48B, 16B-aligned)
#define XCH  (32 * XSTR)  // bf16 per 32x16 chunk
#define SSTR 16           // bf16 per lane-slot in spike chunks (32B)
#define SCH  (32 * SSTR)

// A fragment (16x32 bf16, MxK) from padded row-major LDS w1 at (m0, k0).
// Lane l: M = l&15, K chunks {hi*16..+7} and {hi*16+8..+15} -> two b128 loads.
__device__ __forceinline__ v16bf load_a_lds(const bf16* base, int m, int hi) {
  const bf16* p = base + m * W1S + hi * 16;
  v8bf lo = *(const v8bf*)(p);
  v8bf hh = *(const v8bf*)(p + 8);
  return __builtin_shufflevector(lo, hh, 0,1,2,3,4,5,6,7,8,9,10,11,12,13,14,15);
}

// B fragment (32x16) from a fragment-native chunk: slot s=(K'/16)*16+n' holds
// e=K'%16 contiguously. Lane l reads its own slot: two aligned b128 loads.
__device__ __forceinline__ v16bf load_frag_native(const bf16* chunk, int lane, int str) {
  const bf16* p = chunk + lane * str;
  v8bf lo = *(const v8bf*)(p);
  v8bf hh = *(const v8bf*)(p + 8);
  return __builtin_shufflevector(lo, hh, 0,1,2,3,4,5,6,7,8,9,10,11,12,13,14,15);
}

__global__ void __launch_bounds__(256)
snn_fused_kernel(const float* __restrict__ x,
                 const float* __restrict__ w1,
                 const float* __restrict__ bn1_g, const float* __restrict__ bn1_b,
                 const float* __restrict__ bn1_m, const float* __restrict__ bn1_v,
                 const float* __restrict__ w2,
                 const float* __restrict__ bn2_g, const float* __restrict__ bn2_b,
                 const float* __restrict__ bn2_m, const float* __restrict__ bn2_v,
                 float* __restrict__ out)
{
  __shared__ bf16  lds_w1[HID_ * W1S];      // 65 KB: w1 bf16, padded rows
  __shared__ bf16  lds_xf[32 * XCH];        // 48 KB: half-K x tile, fragment order
  __shared__ bf16  lds_sf[32 * SCH];        // 32 KB: spikes (4 t x 2 ks x 4 nt chunks)
  __shared__ float lds_s1[HID_], lds_sh1[HID_], lds_s2[OUT_], lds_sh2[OUT_];

  const int tid  = threadIdx.x;
  const int lane = tid & 31;
  const int wave = tid >> 5;
  const int nlo  = lane & 15;
  const int hi   = lane >> 4;
  const int b    = blockIdx.y;
  const int n0   = blockIdx.x * NT;

  // Fold BN params.
  if (tid < HID_) {
    float sc = bn1_g[tid] * rsqrtf(bn1_v[tid] + 1e-5f);
    lds_s1[tid]  = sc;
    lds_sh1[tid] = bn1_b[tid] - bn1_m[tid] * sc;
  } else if (tid < HID_ + OUT_) {
    int i = tid - HID_;
    float sc = bn2_g[i] * rsqrtf(bn2_v[i] + 1e-5f);
    lds_s2[i]  = sc;
    lds_sh2[i] = bn2_b[i] - bn2_m[i] * sc;
  }

  // Stage w1 (64x512 fp32) -> bf16 LDS (padded rows).
  {
    const int r  = tid >> 7;          // 0..1
    const int c4 = (tid & 127) * 4;   // 0..508
#pragma unroll 4
    for (int p = 0; p < 32; ++p) {
      int row = p * 2 + r;
      float4 f = *(const float4*)(w1 + row * C_ + c4);
      v4bf v = { (bf16)f.x, (bf16)f.y, (bf16)f.z, (bf16)f.w };
      *(v4bf*)(lds_w1 + row * W1S + c4) = v;
    }
  }
  __syncthreads();

  // GEMM1 tile ownership: wave -> (m-tile, two n-tiles), fixed across t so the
  // LIF membrane state stays in accumulator-shaped registers.
  const int m0 = (wave & 3) * 16;
  const int nA = (wave >> 2) * 32;
  const int nB = nA + 16;
  float vst0[8], vst1[8];
#pragma unroll
  for (int v = 0; v < 8; ++v) { vst0[v] = 0.f; vst1[v] = 0.f; }

  // x staging map: each thread owns one n column and 4 consecutive C rows per
  // pass -> 4 coalesced b32 global loads, one packed b64 LDS store.
  const int sn  = tid & 63;          // n column 0..63
  const int scq = (tid >> 6) * 4;    // c-quad base within 16-row pass group

  for (int t = 0; t < T_; ++t) {
    v8f acc0 = {}; v8f acc1 = {};
    const float* xsrc = x + ((size_t)(t * B_ + b)) * C_ * N_ + n0;
    for (int kk = 0; kk < 2; ++kk) {
      // Stage 256 K-rows of x (fp32 -> bf16) into fragment-native chunks.
#pragma unroll 4
      for (int p = 0; p < 16; ++p) {
        int cl = p * 16 + scq;                       // local K, multiple of 4
        const float* gp = xsrc + (size_t)(kk * 256 + cl) * N_ + sn;
        float f0 = gp[0];
        float f1 = gp[N_];
        float f2 = gp[2 * N_];
        float f3 = gp[3 * N_];
        v4bf v = { (bf16)f0, (bf16)f1, (bf16)f2, (bf16)f3 };
        int k8 = cl >> 5, Kp = cl & 31;
        bf16* dst = lds_xf + (k8 * 4 + (sn >> 4)) * XCH
                           + ((Kp >> 4) * 16 + (sn & 15)) * XSTR + (Kp & 15);
        *(v4bf*)dst = v;
      }
      // Prefetch hint for the next chunk (global_prefetch_b8).
      if (t < T_ - 1 || kk == 0) {
        const float* nxt = (kk == 0) ? (xsrc + (size_t)256 * N_)
                                     : (x + ((size_t)((t + 1) * B_ + b)) * C_ * N_ + n0);
        __builtin_prefetch(nxt + (size_t)scq * N_ + sn, 0, 0);
      }
      __syncthreads();
#pragma unroll
      for (int k8 = 0; k8 < 8; ++k8) {
        v16bf a  = load_a_lds(lds_w1 + m0 * W1S + kk * 256 + k8 * 32, nlo, hi);
        v16bf b0 = load_frag_native(lds_xf + (k8 * 4 + (nA >> 4)) * XCH, lane, XSTR);
        v16bf b1 = load_frag_native(lds_xf + (k8 * 4 + (nB >> 4)) * XCH, lane, XSTR);
        acc0 = __builtin_amdgcn_wmma_f32_16x16x32_bf16(false, a, false, b0,
                                                       (short)0, acc0, false, false);
        acc1 = __builtin_amdgcn_wmma_f32_16x16x32_bf16(false, a, false, b1,
                                                       (short)0, acc1, false, false);
      }
      __syncthreads();
    }
    // BN1 + LIF step; pack 8 spikes per tile and store as one b128.
    v8bf sp0, sp1;
#pragma unroll
    for (int v = 0; v < 8; ++v) {
      int   m  = m0 + v + hi * 8;
      float sc = lds_s1[m], sh = lds_sh1[m];
      float h0 = acc0[v] * sc + sh;
      float h1 = acc1[v] * sc + sh;
      float va = vst0[v] + (h0 - vst0[v]) * 0.5f;   // tau = 2
      float vb = vst1[v] + (h1 - vst1[v]) * 0.5f;
      float sa = (va >= 1.0f) ? 1.0f : 0.0f;        // heaviside(v - v_th)
      float sb = (vb >= 1.0f) ? 1.0f : 0.0f;
      sp0[v] = (bf16)sa;
      sp1[v] = (bf16)sb;
      vst0[v] = va * (1.0f - sa);                   // hard reset
      vst1[v] = vb * (1.0f - sb);
    }
    {
      const int ks  = m0 >> 5;            // K-step index of this m-tile
      const int hi2 = (m0 >> 4) & 1;      // slot half within the 32-K chunk
      bf16* d0 = lds_sf + ((t * 2 + ks) * 4 + (nA >> 4)) * SCH
                        + (hi2 * 16 + nlo) * SSTR + hi * 8;
      bf16* d1 = lds_sf + ((t * 2 + ks) * 4 + (nB >> 4)) * SCH
                        + (hi2 * 16 + nlo) * SSTR + hi * 8;
      *(v8bf*)d0 = sp0;
      *(v8bf*)d1 = sp1;
    }
  }
  __syncthreads();

  // GEMM2: wave -> (t, p-tile); y[p][n] = sum_o w2[p][o] * s[o][n].
  const int t2  = wave >> 1;
  const int m02 = (wave & 1) * 16;
  v8f acc2[4] = {};
#pragma unroll
  for (int ks = 0; ks < 2; ++ks) {
    const int k = ks * 32;
    // A fragment of w2 straight from global fp32 (tiny, L2-resident).
    const float* pw = w2 + (m02 + nlo) * HID_ + k + hi * 16;
    v16bf a;
#pragma unroll
    for (int e = 0; e < 8; ++e) { a[e] = (bf16)pw[e]; a[8 + e] = (bf16)pw[8 + e]; }
#pragma unroll
    for (int nt = 0; nt < 4; ++nt) {
      v16bf bs = load_frag_native(lds_sf + ((t2 * 2 + ks) * 4 + nt) * SCH, lane, SSTR);
      acc2[nt] = __builtin_amdgcn_wmma_f32_16x16x32_bf16(false, a, false, bs,
                                                         (short)0, acc2[nt], false, false);
    }
  }
  // BN2 + rearranged store: y[t,b,p,n], p = cg*8+l -> out[cg,t,b,l,h,w].
#pragma unroll
  for (int nt = 0; nt < 4; ++nt) {
#pragma unroll
    for (int v = 0; v < 8; ++v) {
      int   p   = m02 + v + hi * 8;
      float val = acc2[nt][v] * lds_s2[p] + lds_sh2[p];
      int   cg  = p >> 3, l = p & 7;
      int   n   = n0 + nt * 16 + nlo;
      out[(((size_t)(cg * T_ + t2) * B_ + b) * 8 + l) * N_ + n] = val;
    }
  }
}

extern "C" void kernel_launch(void* const* d_in, const int* in_sizes, int n_in,
                              void* d_out, int out_size, void* d_ws, size_t ws_size,
                              hipStream_t stream) {
  (void)in_sizes; (void)n_in; (void)out_size; (void)d_ws; (void)ws_size;
  const float* x     = (const float*)d_in[0];
  const float* w1    = (const float*)d_in[1];
  const float* bn1_g = (const float*)d_in[2];
  const float* bn1_b = (const float*)d_in[3];
  const float* bn1_m = (const float*)d_in[4];
  const float* bn1_v = (const float*)d_in[5];
  const float* w2    = (const float*)d_in[6];
  const float* bn2_g = (const float*)d_in[7];
  const float* bn2_b = (const float*)d_in[8];
  const float* bn2_m = (const float*)d_in[9];
  const float* bn2_v = (const float*)d_in[10];
  float* out = (float*)d_out;

  dim3 grid(N_ / NT, B_);   // 16 x 32 = 512 blocks
  dim3 block(256);          // 8 waves
  snn_fused_kernel<<<grid, block, 0, stream>>>(x, w1, bn1_g, bn1_b, bn1_m, bn1_v,
                                               w2, bn2_g, bn2_b, bn2_m, bn2_v, out);
}
